// s2cnn_equators_16475494547637
// MI455X (gfx1250) — matrix operand order repaired and checked by reading, not compiled, not run on hardware
//
#include <hip/hip_runtime.h>
#include <math.h>
#include <stdint.h>

// ---------------- types ----------------
typedef _Float16 v16h __attribute__((ext_vector_type(16)));
typedef float    v8f  __attribute__((ext_vector_type(8)));
typedef unsigned int u32x4 __attribute__((ext_vector_type(4)));
typedef int          i32x8 __attribute__((ext_vector_type(8)));
typedef int          i32x4 __attribute__((ext_vector_type(4)));

#define DEVFN static __device__ __forceinline__

// ---------------- problem dims ----------------
constexpr int BATCH = 8, F0 = 12, F1 = 32, F2 = 64;
constexpr int LM = 16;                       // l = 0..15 (b_out = 16 for both convs)
constexpr float S2SCALE  = 0.03608439182435161f;  // 1/sqrt(2*32*12)
constexpr float SO3SCALE = 0.03125f;              // 1/sqrt(2*16*32)
constexpr long X1N = 8388608L;               // 8*32*32^3
constexpr long X2N = 16777216L;              // 8*64*32^3

// ---------------- workspace layout (float offsets) ----------------
constexpr long OFF_QW64  = 0;
constexpr long OFF_QW32  = 64;
constexpr long OFF_DPI2  = 128;                         // 16 * 1024 (31x31 padded per l)
constexpr long OFF_D64   = OFF_DPI2 + 16L*1024;         // 16 * 64 betas * 1024
constexpr long OFF_D32   = OFF_D64  + 16L*64*1024;      // 16 * 32 betas * 1024
constexpr long OFF_WF1   = OFF_D32  + 16L*32*1024;      // 2 * 12*32*64
constexpr long OFF_WF2   = OFF_WF1  + 2L*12*32*64;      // 2 * 32*64*32
constexpr long OFF_XF    = OFF_WF2  + 2L*32*64*32;      // 2 * 8*12*64*64
constexpr long OFF_XL1   = OFF_XF   + 2L*8*12*64*64;    // 2 * 8*12*32
constexpr long OFF_ZL1   = OFF_XL1  + 2L*8*12*32;       // 2 * 8*32*32*32
constexpr long OFF_FREQ1 = OFF_ZL1  + 2L*8*32*32*32;    // 2 * X1N
constexpr long OFF_XF2   = OFF_XF;                      // reused after conv1 finishes
constexpr long OFF_XL2   = OFF_XF2  + 2L*X1N;           // 2 * 8*32*1024
constexpr long OFF_YL2   = OFF_XL2  + 2L*8*32*1024;     // 2 * 1024*2048
constexpr long OFF_ZL2   = OFF_YL2  + 2L*1024*2048;     // 2 * 8*32*2048
constexpr long OFF_FREQ2 = OFF_ZL2  + 2L*8*32*2048;     // 2 * X2N
// total ~57.9M floats (~232 MB)

// ---------------- WMMA helpers (wave32; 16x16x32 f16 -> f32 acc) ----------------
DEVFN int kmapA(int lane, int h) { return ((lane >> 4) & 1) * 16 + h; }
DEVFN int kmapB(int lane, int h) { return ((lane >> 4) & 1) * 16 + h; }

DEVFN v8f wmma16(v16h a, v16h b, v8f c) {
  return __builtin_amdgcn_wmma_f32_16x16x32_f16(false, a, false, b, (short)0, c, false, false);
}

DEVFN v16h hneg(v16h x) {
  v16h r;
#pragma unroll
  for (int h = 0; h < 16; ++h) r[h] = (_Float16)(-(float)x[h]);
  return r;
}

// DFT twiddle fragments, W[a][m] = exp(sgn * i * 2*pi*a*m/32); re=cos, im=sgn*sin
DEVFN void dft_fragA(int lane, int tileRow, float sgn, v16h& fr, v16h& fi) {
  const int a = tileRow + (lane & 15);
  v16h r, im;
#pragma unroll
  for (int h = 0; h < 16; ++h) {
    int m = kmapA(lane, h);
    float s, c;
    __sincosf(0.19634954084936207f * (float)(a * m), &s, &c);
    r[h] = (_Float16)c; im[h] = (_Float16)(sgn * s);
  }
  fr = r; fi = im;
}
DEVFN void dft_fragB(int lane, int tileCol, float sgn, v16h& fr, v16h& fi) {
  const int g = tileCol + (lane & 15);
  v16h r, im;
#pragma unroll
  for (int h = 0; h < 16; ++h) {
    int n = kmapB(lane, h);
    float s, c;
    __sincosf(0.19634954084936207f * (float)(n * g), &s, &c);
    r[h] = (_Float16)c; im[h] = (_Float16)(sgn * s);
  }
  fr = r; fi = im;
}
DEVFN v16h lds_fragB(int lane, int tileCol, const float (*S)[32], int relu) {
  const int n = tileCol + (lane & 15);
  v16h r;
#pragma unroll
  for (int h = 0; h < 16; ++h) {
    float v = S[kmapB(lane, h)][n];
    if (relu) v = fmaxf(v, 0.f);
    r[h] = (_Float16)v;
  }
  return r;
}
DEVFN v16h lds_fragA(int lane, int tileRow, const float (*G)[32]) {
  const int a = tileRow + (lane & 15);
  v16h r;
#pragma unroll
  for (int h = 0; h < 16; ++h) r[h] = (_Float16)G[a][kmapA(lane, h)];
  return r;
}

// ---------------- TDM: async 4KB tile load into LDS (1-D D#) ----------------
#if __has_builtin(__builtin_amdgcn_tensor_load_to_lds) && __has_builtin(__builtin_amdgcn_s_wait_tensorcnt)
#define S2CNN_USE_TDM 1
DEVFN void tdm_load_tile(const float* gsrc, unsigned ldsOff) {
  unsigned long long ga = (unsigned long long)(uintptr_t)gsrc;
  u32x4 g0;
  g0[0] = 1u;                                  // count = 1 valid descriptor
  g0[1] = ldsOff;                              // lds_addr (bytes)
  g0[2] = (unsigned)(ga & 0xffffffffu);        // global_addr[31:0]
  g0[3] = (unsigned)((ga >> 32) & 0x1ffffffu)  // global_addr[56:32]
        | (2u << 30);                          // type = 2
  i32x8 g1;
  g1[0] = (2 << 16);           // data_size = 2 (4-byte elements)
  g1[1] = (int)(1024u << 16);  // tensor_dim0 = 1024 (low 16 at bits 63:48)
  g1[2] = (int)(1u << 16);     // tensor_dim0 hi = 0, tensor_dim1 = 1
  g1[3] = (int)(1024u << 16);  // tensor_dim1 hi = 0, tile_dim0 = 1024
  g1[4] = 1;                   // tile_dim1 = 1, tile_dim2 = 0
  g1[5] = 1024;                // tensor_dim0_stride (low 32)
  g1[6] = (int)(1024u << 16);  // stride0 hi = 0, tensor_dim1_stride low16
  g1[7] = 0;
  i32x4 g2 = {0, 0, 0, 0};
  i32x4 g3 = {0, 0, 0, 0};
  i32x8 g4 = {0, 0, 0, 0, 0, 0, 0, 0};
  // 6-arg form (clang-23 / therock-10.0 headers)
  __builtin_amdgcn_tensor_load_to_lds(g0, g1, g2, g3, g4, 0);
}
#endif

// ---------------- misc small kernels ----------------
__global__ void zero_kernel(float* p, long n) {
  long i = (long)blockIdx.x * blockDim.x + threadIdx.x;
  long stride = (long)gridDim.x * blockDim.x;
  for (; i < n; i += stride) p[i] = 0.f;
}

__global__ void qw_kernel(float* qw64, float* qw32) {
  int t = blockIdx.x * blockDim.x + threadIdx.x;
  if (t >= 96) return;
  int bq = (t < 64) ? 32 : 16;
  int z = (t < 64) ? t : t - 64;
  double theta = 3.14159265358979323846 * (2.0 * z + 1.0) / (4.0 * bq);
  double s = 0.0;
  for (int j = 0; j < bq; ++j) s += sin(theta * (2 * j + 1)) / (2.0 * j + 1.0);
  double q = 2.0 / bq * sin(theta) * s;
  if (t < 64) qw64[z] = (float)q; else qw32[z] = (float)q;
}

DEVFN double ipow(double x, int n) { double r = 1.0; for (int i = 0; i < n; ++i) r *= x; return r; }

// Wigner small-d via l-recursion (interior 3-term recurrence + closed-form +-l borders).
// Tables: d^l stored row-major (m' row, m col), 31x31 region padded to 1024 floats per (l, beta).
__global__ void wigner_kernel(float* d64t, float* d32t, float* dpi2t) {
  int t = blockIdx.x * blockDim.x + threadIdx.x;
  if (t >= 97) return;
  float* base; int nb; int z; double beta;
  if (t < 64)      { base = d64t;  nb = 64; z = t;      beta = 3.14159265358979323846 * (2.0 * z + 1.0) / 128.0; }
  else if (t < 96) { base = d32t;  nb = 32; z = t - 64; beta = 3.14159265358979323846 * (2.0 * z + 1.0) / 64.0; }
  else             { base = dpi2t; nb = 1;  z = 0;      beta = 3.14159265358979323846 * 0.5; }
  double cb = cos(beta), sb = sin(beta);
  double chb = cos(0.5 * beta), shb = sin(0.5 * beta);
  const double r2 = 0.7071067811865475244;
  base[(0 * nb + z) * 1024] = 1.0f;                     // l = 0
  {                                                     // l = 1
    float* p = base + (long)(1 * nb + z) * 1024;
    p[0 * 31 + 0] = (float)(0.5 * (1 + cb)); p[0 * 31 + 1] = (float)(sb * r2);  p[0 * 31 + 2] = (float)(0.5 * (1 - cb));
    p[1 * 31 + 0] = (float)(-sb * r2);       p[1 * 31 + 1] = (float)cb;         p[1 * 31 + 2] = (float)(sb * r2);
    p[2 * 31 + 0] = (float)(0.5 * (1 - cb)); p[2 * 31 + 1] = (float)(-sb * r2); p[2 * 31 + 2] = (float)(0.5 * (1 + cb));
  }
  for (int l = 2; l < LM; ++l) {
    float* d        = base + (long)(l * nb + z) * 1024;
    const float* d1 = base + (long)((l - 1) * nb + z) * 1024;
    const float* d2 = base + (long)((l - 2) * nb + z) * 1024;
    double L = (double)l;
    for (int mp = -(l - 1); mp <= l - 1; ++mp) {
      for (int mm = -(l - 1); mm <= l - 1; ++mm) {
        double u = L * (2 * L - 1) / sqrt((L * L - mp * mp) * (L * L - mm * mm));
        double vv = cb - (double)(mp * mm) / (L * (L - 1));
        double w = sqrt(((L - 1) * (L - 1) - mp * mp) * ((L - 1) * (L - 1) - mm * mm)) / ((L - 1) * (2 * L - 1));
        double p1 = d1[(mp + l - 1) * 31 + (mm + l - 1)];
        double p2 = (mp >= -(l - 2) && mp <= l - 2 && mm >= -(l - 2) && mm <= l - 2)
                        ? (double)d2[(mp + l - 2) * 31 + (mm + l - 2)] : 0.0;
        d[(mp + l) * 31 + (mm + l)] = (float)(u * (vv * p1 - w * p2));
      }
    }
    double Am = 1.0;   // sqrt(C(2l, l+mm)), built incrementally
    for (int mm = -l; mm <= l; ++mm) {
      double cp = ipow(chb, l + mm), cn2 = ipow(chb, l - mm);
      double sp = ipow(shb, l + mm), sn  = ipow(shb, l - mm);
      double sgnLm = ((l - mm) & 1) ? -1.0 : 1.0;   // (-shb)^(l-m)
      double sgnLp = ((l + mm) & 1) ? -1.0 : 1.0;   // (-shb)^(l+m)
      d[(2 * l) * 31 + (mm + l)] = (float)(Am * cp * sn * sgnLm);   // row m' = +l
      d[0 * 31 + (mm + l)]       = (float)(Am * cn2 * sp);          // row m' = -l
      d[(mm + l) * 31 + 2 * l]   = (float)(Am * cp * sn);           // col m  = +l
      d[(mm + l) * 31 + 0]       = (float)(Am * cn2 * sp * sgnLp);  // col m  = -l
      if (mm < l) Am *= sqrt((double)(l - mm) / (double)(l + mm + 1));
    }
  }
}

// xf[b,i,z,m] = (1/64) sum_a x[b,i,z,a] e^{-2pi i m a/64}
__global__ void fft1_kernel(const float* __restrict__ x, float* __restrict__ xre, float* __restrict__ xim) {
  int t = blockIdx.x * blockDim.x + threadIdx.x;
  if (t >= BATCH * F0 * 64 * 64) return;
  int m = t & 63; int biz = t >> 6;
  const float* row = x + (long)biz * 64;
  const float c = -6.2831853071795864769f / 64.f;
  float sr = 0.f, si = 0.f;
  for (int a = 0; a < 64; ++a) {
    float s, co; __sincosf(c * (float)(m * a), &s, &co);
    sr += row[a] * co; si += row[a] * s;
  }
  xre[t] = sr * (1.f / 64.f); xim[t] = si * (1.f / 64.f);
}

// wre/wim[r,k] = sum_j w[r,j] e^{-2pi i j k / n}
__global__ void dftw_kernel(const float* __restrict__ w, float* __restrict__ wre,
                            float* __restrict__ wim, int rows, int n) {
  int t = blockIdx.x * blockDim.x + threadIdx.x;
  if (t >= rows * n) return;
  int k = t % n, r = t / n;
  const float* src = w + (long)r * n;
  const float c = -6.2831853071795864769f / (float)n;
  float sr = 0.f, si = 0.f;
  for (int j = 0; j < n; ++j) {
    float s, co; __sincosf(c * (float)(j * k), &s, &co);
    sr += src[j] * co; si += src[j] * s;
  }
  wre[t] = sr; wim[t] = si;
}

// ---------------- S2 conv, per-l stages ----------------
__global__ void s2_xl_kernel(const float* qw64, const float* D64,
                             const float* xre, const float* xim,
                             float* xlre, float* xlim, int l) {
  int Nl = 2 * l + 1;
  int t = blockIdx.x * blockDim.x + threadIdx.x;
  if (t >= BATCH * F0 * Nl) return;
  int m = t % Nl; int bi = t / Nl;
  int col = (m - l + 64) & 63;
  float sr = 0.f, si = 0.f;
  for (int z = 0; z < 64; ++z) {
    float dv = qw64[z] * D64[((long)(l * 64 + z)) * 1024 + m * 31 + l];
    long xi = ((long)(bi * 64 + z)) * 64 + col;
    sr += dv * xre[xi]; si += dv * xim[xi];
  }
  xlre[bi * 32 + m] = sr; xlim[bi * 32 + m] = si;
}

__global__ void s2_z_kernel(const float* DPI2, const float* wre, const float* wim,
                            const float* xlre, const float* xlim,
                            float* zre, float* zim, int l) {
  int Nl = 2 * l + 1;
  int t = blockIdx.x * blockDim.x + threadIdx.x;
  if (t >= BATCH * F1 * Nl * Nl) return;
  int n = t % Nl; int m = (t / Nl) % Nl; int o = (t / (Nl * Nl)) % F1; int b = t / (Nl * Nl * F1);
  int ncol = (n - l + 64) & 63;
  float dp = S2SCALE * DPI2[(long)l * 1024 + n * 31 + l];
  float ar = 0.f, ai = 0.f;
  for (int i = 0; i < F0; ++i) {
    float xr = xlre[(b * F0 + i) * 32 + m], xi = xlim[(b * F0 + i) * 32 + m];
    float yr = wre[(long)(i * F1 + o) * 64 + ncol], yi = wim[(long)(i * F1 + o) * 64 + ncol];
    ar += xr * yr - xi * yi;
    ai += xr * yi + xi * yr;
  }
  long oi = (((long)(b * F1 + o)) * 32 + m) * 32 + n;
  zre[oi] = dp * ar; zim[oi] = dp * ai;
}

__global__ void s2_scatter_kernel(const float* D32, const float* zre, const float* zim,
                                  float* fre, float* fim, int l) {
  int Nl = 2 * l + 1;
  long t = (long)blockIdx.x * blockDim.x + threadIdx.x;
  long total = (long)BATCH * F1 * 32 * Nl * Nl;
  if (t >= total) return;
  int n = (int)(t % Nl); int m = (int)((t / Nl) % Nl); int z = (int)((t / ((long)Nl * Nl)) % 32);
  int o = (int)((t / ((long)Nl * Nl * 32)) % F1); int b = (int)(t / ((long)Nl * Nl * 32 * F1));
  float d = (float)(2 * l + 1) * D32[((long)(l * 32 + z)) * 1024 + m * 31 + n];
  int mo = (m - l + 32) & 31, no = (n - l + 32) & 31;
  long fi = (((long)(b * F1 + o)) * 32 + z) * 1024 + mo * 32 + no;
  long zi = (((long)(b * F1 + o)) * 32 + m) * 32 + n;
  fre[fi] += d * zre[zi];
  fim[fi] += d * zim[zi];
}

// ---------------- fused 32x32 2D transform (WMMA + TDM + LDS) ----------------
// T = W * S * W  (W symmetric DFT matrix, W[a][m] = exp(sgn*i*2pi*a*m/32)).
// in_im==null  : real input (optionally ReLU'd)          -> S imag = 0
// out_im==null : real output = Re(T)*outScale + bias[ch] ; else complex*outScale
__global__ void __launch_bounds__(128)
transform2d_kernel(const float* __restrict__ in_re, const float* __restrict__ in_im,
                   float* __restrict__ out_re, float* __restrict__ out_im,
                   const float* __restrict__ bias, int nch,
                   float sgn, float outScale, int reluIn) {
  __shared__ float Sr[32][32];
  __shared__ float Si[32][32];
  __shared__ float Gr[32][32];
  __shared__ float Gi[32][32];

  const int tid = threadIdx.x, wid = tid >> 5, lane = tid & 31;
  const long slice = blockIdx.x;
  const bool cplxIn = (in_im != nullptr);
  const bool cplxOut = (out_im != nullptr);
  const float* src_re = in_re + slice * 1024;
  const float* src_im = cplxIn ? (in_im + slice * 1024) : nullptr;

#ifdef S2CNN_USE_TDM
  if (wid == 0) {   // wave-uniform branch: only wave 0 issues the DMAs
    tdm_load_tile(src_re, (unsigned)(uintptr_t)&Sr[0][0]);
    if (cplxIn) tdm_load_tile(src_im, (unsigned)(uintptr_t)&Si[0][0]);
    __builtin_amdgcn_s_wait_tensorcnt(0);
  }
#else
  for (int i = tid; i < 1024; i += 128) {
    Sr[i >> 5][i & 31] = src_re[i];
    if (cplxIn) Si[i >> 5][i & 31] = src_im[i];
  }
#endif
  __syncthreads();

  // stage 1: G = W * S   (one wave per 16x16 tile; full K=32 in one wmma)
  {
    const int ta = (wid & 1) * 16;
    const int tn = (wid >> 1) * 16;
    v16h wr, wi; dft_fragA(lane, ta, sgn, wr, wi);
    v16h sbr = lds_fragB(lane, tn, Sr, reluIn);
    v8f accR = {}; v8f accI = {};
    if (cplxIn) {
      v16h sbi = lds_fragB(lane, tn, Si, 0);
      accR = wmma16(hneg(wi), sbi, accR);
      accI = wmma16(wr, sbi, accI);
    }
    accR = wmma16(wr, sbr, accR);
    accI = wmma16(wi, sbr, accI);
    const int col = tn + (lane & 15);
    const int row0 = ta + ((lane >> 4) & 1) * 8;
#pragma unroll
    for (int v = 0; v < 8; ++v) { Gr[row0 + v][col] = accR[v]; Gi[row0 + v][col] = accI[v]; }
  }
  __syncthreads();

  // stage 2: T = G * W
  {
    const int ta = (wid & 1) * 16;
    const int tg = (wid >> 1) * 16;
    v16h gar = lds_fragA(lane, ta, Gr);
    v16h gai = lds_fragA(lane, ta, Gi);
    v16h w2r, w2i; dft_fragB(lane, tg, sgn, w2r, w2i);
    v8f tr = {};
    tr = wmma16(gai, hneg(w2i), tr);
    tr = wmma16(gar, w2r, tr);
    const int col = tg + (lane & 15);
    const int row0 = ta + ((lane >> 4) & 1) * 8;
    if (cplxOut) {
      v8f ti = {};
      ti = wmma16(gai, w2r, ti);
      ti = wmma16(gar, w2i, ti);
#pragma unroll
      for (int v = 0; v < 8; ++v) {
        long o = slice * 1024 + (row0 + v) * 32 + col;
        out_re[o] = tr[v] * outScale;
        out_im[o] = ti[v] * outScale;
      }
    } else {
      const float bv = bias ? bias[(int)((slice >> 5) % nch)] : 0.f;
#pragma unroll
      for (int v = 0; v < 8; ++v) {
        long o = slice * 1024 + (row0 + v) * 32 + col;
        out_re[o] = tr[v] * outScale + bv;
      }
    }
  }
}

// ---------------- SO3 conv per-l stages ----------------
// A[b][m][i*kpad+k] = Xl2[b,i,m,k] ; zero-padded
__global__ void so3_xl_kernel(const float* qw32, const float* D32,
                              const float* xre, const float* xim,
                              float* are, float* aim, int l, int kpad) {
  int Nl = 2 * l + 1;
  int t = blockIdx.x * blockDim.x + threadIdx.x;
  if (t >= BATCH * 32 * 32 * kpad) return;
  int k = t % kpad; int m = (t / kpad) & 31; int i = (t / (kpad * 32)) & 31; int b = t / (kpad * 1024);
  long oi = ((long)(b * 32 + m)) * 1024 + i * kpad + k;
  if (m >= Nl || k >= Nl) { are[oi] = 0.f; aim[oi] = 0.f; return; }
  int cm = (m - l + 32) & 31, cn = (k - l + 32) & 31;
  float sr = 0.f, si = 0.f;
  for (int z = 0; z < 32; ++z) {
    float dv = qw32[z] * D32[((long)(l * 32 + z)) * 1024 + m * 31 + k];
    long xi = (((long)(b * 32 + i)) * 32 + z) * 1024 + cm * 32 + cn;
    sr += dv * xre[xi]; si += dv * xim[xi];
  }
  are[oi] = sr; aim[oi] = si;
}

// B[i*kpad+k][o*32+n] = SO3SCALE * dpi2[k,n] * Wf2[i,o,(k-l) mod 32] ; zero-padded
__global__ void so3_y_kernel(const float* DPI2, const float* wre, const float* wim,
                             float* bre, float* bim, int l, int kpad) {
  int Nl = 2 * l + 1;
  long t = (long)blockIdx.x * blockDim.x + threadIdx.x;
  long total = (long)32 * kpad * 64 * 32;
  if (t >= total) return;
  int n = (int)(t & 31); int o = (int)((t >> 5) & 63); long rest = t >> 11;
  int k = (int)(rest % kpad); int i = (int)(rest / kpad);
  long oi = ((long)(i * kpad + k)) * 2048 + o * 32 + n;
  if (k >= Nl || n >= Nl) { bre[oi] = 0.f; bim[oi] = 0.f; return; }
  int ck = (k - l + 32) & 31;
  float dp = SO3SCALE * DPI2[(long)l * 1024 + k * 31 + n];
  long wi2 = ((long)(i * F2 + o)) * 32 + ck;
  bre[oi] = dp * wre[wi2];
  bim[oi] = dp * wim[wi2];
}

// C[b][m][on] = sum_{ik} A[b][m][ik] * B[ik][on]   (complex, WMMA f16->f32)
__global__ void __launch_bounds__(32)
so3_zmm_kernel(const float* __restrict__ Are, const float* __restrict__ Aim,
               const float* __restrict__ Bre, const float* __restrict__ Bim,
               float* __restrict__ Cre, float* __restrict__ Cim, int ktot) {
  const int lane = threadIdx.x;
  const int ot = blockIdx.x & 127;
  const int mt = (blockIdx.x >> 7) & 1;
  const int b = blockIdx.x >> 8;
  const long arow = ((long)(b * 32 + mt * 16 + (lane & 15))) * 1024;
  const int bcol = ot * 16 + (lane & 15);
  v8f prr = {}, pii = {}, pri = {}, pir = {};
  for (int k0 = 0; k0 < ktot; k0 += 32) {
    v16h ar, ai, br, bi;
#pragma unroll
    for (int h = 0; h < 16; ++h) {
      int ka = k0 + kmapA(lane, h);
      ar[h] = (_Float16)Are[arow + ka];
      ai[h] = (_Float16)Aim[arow + ka];
      int kb = k0 + kmapB(lane, h);
      br[h] = (_Float16)Bre[(long)kb * 2048 + bcol];
      bi[h] = (_Float16)Bim[(long)kb * 2048 + bcol];
    }
    prr = wmma16(ar, br, prr);
    pii = wmma16(ai, bi, pii);
    pri = wmma16(ar, bi, pri);
    pir = wmma16(ai, br, pir);
  }
  v8f cr = prr - pii;
  v8f ci = pri + pir;
  const int row0 = b * 32 + mt * 16 + ((lane >> 4) & 1) * 8;
#pragma unroll
  for (int v = 0; v < 8; ++v) {
    long o = (long)(row0 + v) * 2048 + bcol;
    Cre[o] = cr[v];
    Cim[o] = ci[v];
  }
}

__global__ void so3_scatter_kernel(const float* D32, const float* zre, const float* zim,
                                   float* fre, float* fim, int l) {
  int Nl = 2 * l + 1;
  long t = (long)blockIdx.x * blockDim.x + threadIdx.x;
  long total = (long)BATCH * F2 * 32 * Nl * Nl;
  if (t >= total) return;
  int n = (int)(t % Nl); int m = (int)((t / Nl) % Nl); int z = (int)((t / ((long)Nl * Nl)) % 32);
  int o = (int)((t / ((long)Nl * Nl * 32)) % F2); int b = (int)(t / ((long)Nl * Nl * 32 * F2));
  float d = (float)(2 * l + 1) * D32[((long)(l * 32 + z)) * 1024 + m * 31 + n];
  int mo = (m - l + 32) & 31, no = (n - l + 32) & 31;
  long fi = (((long)(b * F2 + o)) * 32 + z) * 1024 + mo * 32 + no;
  long zi = ((long)(b * 32 + m)) * 2048 + o * 32 + n;
  fre[fi] += d * zre[zi];
  fim[fi] += d * zim[zi];
}

// ---------------- host side ----------------
static inline unsigned cdiv(long a, long b) { return (unsigned)((a + b - 1) / b); }

extern "C" void kernel_launch(void* const* d_in, const int* in_sizes, int n_in,
                              void* d_out, int out_size, void* d_ws, size_t ws_size,
                              hipStream_t stream) {
  (void)in_sizes; (void)n_in; (void)out_size; (void)ws_size;
  const float* x     = (const float*)d_in[0];
  const float* w_s2  = (const float*)d_in[1];
  const float* b_s2  = (const float*)d_in[2];
  const float* w_so3 = (const float*)d_in[3];
  const float* b_so3 = (const float*)d_in[4];
  float* out = (float*)d_out;
  float* ws = (float*)d_ws;

  float* qw64 = ws + OFF_QW64;
  float* qw32 = ws + OFF_QW32;
  float* dpi2 = ws + OFF_DPI2;
  float* d64 = ws + OFF_D64;
  float* d32 = ws + OFF_D32;
  float* wf1re = ws + OFF_WF1;   float* wf1im = wf1re + (long)12 * 32 * 64;
  float* wf2re = ws + OFF_WF2;   float* wf2im = wf2re + (long)32 * 64 * 32;
  float* xfre = ws + OFF_XF;     float* xfim = xfre + (long)8 * 12 * 64 * 64;
  float* xl1re = ws + OFF_XL1;   float* xl1im = xl1re + (long)8 * 12 * 32;
  float* zl1re = ws + OFF_ZL1;   float* zl1im = zl1re + (long)8 * 32 * 32 * 32;
  float* f1re = ws + OFF_FREQ1;  float* f1im = f1re + X1N;
  float* xf2re = ws + OFF_XF2;   float* xf2im = xf2re + X1N;
  float* xl2re = ws + OFF_XL2;   float* xl2im = xl2re + (long)8 * 32 * 1024;
  float* yl2re = ws + OFF_YL2;   float* yl2im = yl2re + (long)1024 * 2048;
  float* zl2re = ws + OFF_ZL2;   float* zl2im = zl2re + (long)8 * 32 * 2048;
  float* f2re = ws + OFF_FREQ2;  float* f2im = f2re + X2N;

  // constants + input spectra
  qw_kernel<<<1, 128, 0, stream>>>(qw64, qw32);
  wigner_kernel<<<1, 128, 0, stream>>>(d64, d32, dpi2);
  fft1_kernel<<<cdiv((long)8 * 12 * 64 * 64, 256), 256, 0, stream>>>(x, xfre, xfim);
  dftw_kernel<<<cdiv((long)12 * 32 * 64, 256), 256, 0, stream>>>(w_s2, wf1re, wf1im, 12 * 32, 64);
  dftw_kernel<<<cdiv((long)32 * 64 * 32, 256), 256, 0, stream>>>(w_so3, wf2re, wf2im, 32 * 64, 32);
  zero_kernel<<<2048, 256, 0, stream>>>(f1re, 2 * X1N);

  // ---- S2 conv: build freq1 over l ----
  for (int l = 0; l < LM; ++l) {
    int Nl = 2 * l + 1;
    s2_xl_kernel<<<cdiv((long)8 * 12 * Nl, 256), 256, 0, stream>>>(qw64, d64, xfre, xfim, xl1re, xl1im, l);
    s2_z_kernel<<<cdiv((long)8 * 32 * Nl * Nl, 256), 256, 0, stream>>>(dpi2, wf1re, wf1im, xl1re, xl1im, zl1re, zl1im, l);
    s2_scatter_kernel<<<cdiv((long)8 * 32 * 32 * Nl * Nl, 256), 256, 0, stream>>>(d32, zl1re, zl1im, f1re, f1im, l);
  }
  // x1 = Re(conj(F) * freq1 * conj(F)) + b_s2   (ifft2 * no^2)
  transform2d_kernel<<<8192, 128, 0, stream>>>(f1re, f1im, out, nullptr, b_s2, 32, 1.f, 1.f, 0);
  // xf2 = fft2(relu(x1)) / 1024
  transform2d_kernel<<<8192, 128, 0, stream>>>(out, nullptr, xf2re, xf2im, nullptr, 1, -1.f, 1.f / 1024.f, 1);

  // ---- SO3 conv: build freq2 over l ----
  zero_kernel<<<2048, 256, 0, stream>>>(f2re, 2 * X2N);
  for (int l = 0; l < LM; ++l) {
    int Nl = 2 * l + 1;
    int kpad = (Nl + 3) & ~3;
    so3_xl_kernel<<<cdiv((long)8 * 32 * 32 * kpad, 256), 256, 0, stream>>>(qw32, d32, xf2re, xf2im, xl2re, xl2im, l, kpad);
    so3_y_kernel<<<cdiv((long)32 * kpad * 64 * 32, 256), 256, 0, stream>>>(dpi2, wf2re, wf2im, yl2re, yl2im, l, kpad);
    so3_zmm_kernel<<<2048, 32, 0, stream>>>(xl2re, xl2im, yl2re, yl2im, zl2re, zl2im, 32 * kpad);
    so3_scatter_kernel<<<cdiv((long)8 * 64 * 32 * Nl * Nl, 256), 256, 0, stream>>>(d32, zl2re, zl2im, f2re, f2im, l);
  }
  // x2 = Re(conj(F) * freq2 * conj(F)) + b_so3
  transform2d_kernel<<<16384, 128, 0, stream>>>(f2re, f2im, out + X1N, nullptr, b_so3, 64, 1.f, 1.f, 0);
}
